// EGNN_30494267801867
// MI455X (gfx1250) — compile-verified, matrix operand order checked by reading, not run on hardware
//
#include <hip/hip_runtime.h>
#include <hip/hip_bf16.h>
#include <math.h>

// Problem constants (fixed by the reference)
#define HN      256     // hidden size H
#define IN_DIM  16
#define NNODE   4096
#define NEDGE   65536
#define IE_REAL 514     // 2H + 2
#define IEP     544     // 514 padded to 17*32 for K-tiling
#define K2H     512     // node-MLP input K (2H), already 16*32
#define MBLK    32      // edges/nodes per block (2 WMMA M-tiles)

typedef __attribute__((ext_vector_type(16))) _Float16 v16h;
typedef __attribute__((ext_vector_type(8)))  _Float16 v8h;
typedef __attribute__((ext_vector_type(8)))  float    v8f;

__device__ __forceinline__ float silu_f(float t) {
  return t / (1.0f + __expf(-t));
}

// Load a 16x32 f16 WMMA operand fragment from a row-major [16][ldk] tile.
// CDNA5 16-bit A layout: lanes 0-15 row M=lane, K {k0..k0+7, k0+16..k0+23};
// lanes 16-31 row M=lane-16, K {k0+8..k0+15, k0+24..k0+31}.
// Storing W^T (N-major) makes the B fragment load identical.
__device__ __forceinline__ v16h load_frag(const _Float16* __restrict__ base, int ldk) {
  const int lane = threadIdx.x & 31;
  const _Float16* p = base + (lane & 15) * ldk + ((lane >> 4) << 3);
  v8h lo = *(const v8h*)(p);
  v8h hi = *(const v8h*)(p + 16);
  v16h r;
#pragma unroll
  for (int i = 0; i < 8; ++i) { r[i] = lo[i]; r[i + 8] = hi[i]; }
  return r;
}

__device__ __forceinline__ v8f wmma_f16(v16h a, v16h b, v8f c) {
  return __builtin_amdgcn_wmma_f32_16x16x32_f16(false, a, false, b, (short)0, c,
                                                false, false);
}

// ---------------------------------------------------------------------------
// Weight repack: fp32 [G][K][HN] -> fp16 transposed+padded [G][HN][Kpad]
// ---------------------------------------------------------------------------
__global__ void wconv_kernel(const float* __restrict__ src, _Float16* __restrict__ dst,
                             int K, int Kpad, int G) {
  int idx = blockIdx.x * 256 + threadIdx.x;
  int total = G * HN * Kpad;
  if (idx >= total) return;
  int k = idx % Kpad;
  int t = idx / Kpad;
  int n = t % HN;
  int g = t / HN;
  dst[idx] = (k < K) ? (_Float16)src[((size_t)g * K + k) * HN + n] : (_Float16)0.0f;
}

// ---------------------------------------------------------------------------
// Init: dist0 from initial coords; zero agg; copy x
// ---------------------------------------------------------------------------
__global__ void dist0_kernel(const float* __restrict__ x, const int* __restrict__ rowi,
                             const int* __restrict__ coli, float* __restrict__ dist0) {
  int e = blockIdx.x * 256 + threadIdx.x;
  if (e >= NEDGE) return;
  int r = rowi[e], c = coli[e];
  float dx = x[r * 3 + 0] - x[c * 3 + 0];
  float dy = x[r * 3 + 1] - x[c * 3 + 1];
  float dz = x[r * 3 + 2] - x[c * 3 + 2];
  dist0[e] = dx * dx + dy * dy + dz * dz;
}

__global__ void init_state_kernel(const float* __restrict__ x, float* __restrict__ xw,
                                  float* __restrict__ agg) {
  int idx = blockIdx.x * 256 + threadIdx.x;
  if (idx < NNODE * HN) agg[idx] = 0.0f;
  if (idx < NNODE * 3)  xw[idx] = x[idx];
}

// ---------------------------------------------------------------------------
// Embedding: h[N][16] @ emb_w[16][256] + b  -> hwork fp32 + hf16
// ---------------------------------------------------------------------------
__global__ void embed_kernel(const float* __restrict__ h0, const float* __restrict__ w,
                             const float* __restrict__ b, float* __restrict__ hwork,
                             _Float16* __restrict__ hf16) {
  int idx = blockIdx.x * 256 + threadIdx.x;   // N*H total
  int n = idx >> 8, j = idx & 255;
  float acc = b[j];
#pragma unroll
  for (int k = 0; k < IN_DIM; ++k) acc += h0[n * IN_DIM + k] * w[k * HN + j];
  hwork[idx] = acc;
  hf16[idx] = (_Float16)acc;
}

// ---------------------------------------------------------------------------
// coord2diff: radial + normalized diff from current coords; ea = {radial, dist0}
// ---------------------------------------------------------------------------
__global__ void coord2diff_kernel(const float* __restrict__ xw, const int* __restrict__ rowi,
                                  const int* __restrict__ coli, const float* __restrict__ dist0,
                                  float* __restrict__ ea, float* __restrict__ cd) {
  int e = blockIdx.x * 256 + threadIdx.x;
  if (e >= NEDGE) return;
  int r = rowi[e], c = coli[e];
  float dx = xw[r * 3 + 0] - xw[c * 3 + 0];
  float dy = xw[r * 3 + 1] - xw[c * 3 + 1];
  float dz = xw[r * 3 + 2] - xw[c * 3 + 2];
  float rad = dx * dx + dy * dy + dz * dz;
  float inv = 1.0f / (sqrtf(rad + 1e-8f) + 1.0f);
  ea[e * 2 + 0] = rad;
  ea[e * 2 + 1] = dist0[e];
  cd[e * 3 + 0] = dx * inv;
  cd[e * 3 + 1] = dy * inv;
  cd[e * 3 + 2] = dz * inv;
}

// ---------------------------------------------------------------------------
// Shared staging helper: gather [h_row | h_col | ea | pad] for 32 edges
// 256 threads: 8 threads per edge row, 32 halves (4x v8h) each for row & col.
// ---------------------------------------------------------------------------
__device__ __forceinline__ void stage_edge_rows(
    _Float16* __restrict__ A, int* __restrict__ rIdx,
    const _Float16* __restrict__ hf16, const float* __restrict__ ea,
    const int* __restrict__ rowi, const int* __restrict__ coli, int e0) {
  const int tid = threadIdx.x;
  const int r = tid >> 3;       // 0..31
  const int c = tid & 7;        // 0..7
  const int er = rowi[e0 + r], ec = coli[e0 + r];
  if (c == 0) rIdx[r] = er;
  const _Float16* hr = hf16 + (size_t)er * HN + (c << 5);
  const _Float16* hc = hf16 + (size_t)ec * HN + (c << 5);
  v8h* dst = (v8h*)(A + r * IEP);
#pragma unroll
  for (int i = 0; i < 4; ++i) dst[(c << 2) + i]      = *(const v8h*)(hr + 8 * i);
#pragma unroll
  for (int i = 0; i < 4; ++i) dst[32 + (c << 2) + i] = *(const v8h*)(hc + 8 * i);
  if (c == 0) {
    A[r * IEP + 512] = (_Float16)ea[(e0 + r) * 2 + 0];
    A[r * IEP + 513] = (_Float16)ea[(e0 + r) * 2 + 1];
#pragma unroll
    for (int k = IE_REAL; k < IEP; ++k) A[r * IEP + k] = (_Float16)0.0f;
  }
}

// ---------------------------------------------------------------------------
// Edge MLP (fused 2 layers) + scaled atomic segment-sum into agg
// block = 32 edges x full H. 8 waves, 2 N-tiles x 2 M-tiles per wave.
// ---------------------------------------------------------------------------
__global__ __launch_bounds__(256) void edge_mlp_kernel(
    const _Float16* __restrict__ hf16, const float* __restrict__ ea,
    const int* __restrict__ rowi, const int* __restrict__ coli,
    const _Float16* __restrict__ w1t, const float* __restrict__ b1,
    const _Float16* __restrict__ w2t, const float* __restrict__ b2,
    float* __restrict__ agg) {
  __shared__ _Float16 A[MBLK * IEP];   // gathered [h_row | h_col | ea | pad]
  __shared__ _Float16 M1[MBLK * HN];   // layer-1 output (silu, f16)
  __shared__ int rIdx[MBLK];

  const int e0 = blockIdx.x * MBLK;
  const int tid = threadIdx.x;
  const int wave = tid >> 5, lane = tid & 31;

  stage_edge_rows(A, rIdx, hf16, ea, rowi, coli, e0);
  __syncthreads();

  const int nt0 = wave << 1;
  v8f a00 = {}, a01 = {}, a10 = {}, a11 = {};
  for (int kt = 0; kt < IEP / 32; ++kt) {
    const int k0 = kt << 5;
    if (kt + 1 < IEP / 32)
      __builtin_prefetch(w1t + (size_t)nt0 * 16 * IEP + k0 + 32, 0, 1);
    v16h f0 = load_frag(A + k0, IEP);
    v16h f1 = load_frag(A + 16 * IEP + k0, IEP);
    v16h bA = load_frag(w1t + (size_t)(nt0 + 0) * 16 * IEP + k0, IEP);
    v16h bB = load_frag(w1t + (size_t)(nt0 + 1) * 16 * IEP + k0, IEP);
    a00 = wmma_f16(f0, bA, a00);
    a01 = wmma_f16(f0, bB, a01);
    a10 = wmma_f16(f1, bA, a10);
    a11 = wmma_f16(f1, bB, a11);
  }
  { // SiLU + f16 -> LDS (C layout: M = v + 8*(lane>=16), N = nt*16 + lane%16)
    const int mb  = (lane >> 4) << 3;
    const int nb0 = (nt0 << 4) + (lane & 15);
    const int nb1 = nb0 + 16;
    const float bb0 = b1[nb0], bb1 = b1[nb1];
#pragma unroll
    for (int v = 0; v < 8; ++v) {
      const int m = mb + v;
      M1[m * HN + nb0]        = (_Float16)silu_f(a00[v] + bb0);
      M1[m * HN + nb1]        = (_Float16)silu_f(a01[v] + bb1);
      M1[(16 + m) * HN + nb0] = (_Float16)silu_f(a10[v] + bb0);
      M1[(16 + m) * HN + nb1] = (_Float16)silu_f(a11[v] + bb1);
    }
  }
  __syncthreads();

  v8f d00 = {}, d01 = {}, d10 = {}, d11 = {};
  for (int kt = 0; kt < HN / 32; ++kt) {
    const int k0 = kt << 5;
    v16h f0 = load_frag(M1 + k0, HN);
    v16h f1 = load_frag(M1 + 16 * HN + k0, HN);
    v16h bA = load_frag(w2t + (size_t)(nt0 + 0) * 16 * HN + k0, HN);
    v16h bB = load_frag(w2t + (size_t)(nt0 + 1) * 16 * HN + k0, HN);
    d00 = wmma_f16(f0, bA, d00);
    d01 = wmma_f16(f0, bB, d01);
    d10 = wmma_f16(f1, bA, d10);
    d11 = wmma_f16(f1, bB, d11);
  }
  { // SiLU + scaled atomic scatter (segment_sum / 100)
    const int mb  = (lane >> 4) << 3;
    const int nb0 = (nt0 << 4) + (lane & 15);
    const int nb1 = nb0 + 16;
    const float bb0 = b2[nb0], bb1 = b2[nb1];
#pragma unroll
    for (int v = 0; v < 8; ++v) {
      const int nA = rIdx[mb + v];
      const int nB = rIdx[16 + mb + v];
      atomicAdd(&agg[(size_t)nA * HN + nb0], silu_f(d00[v] + bb0) * 0.01f);
      atomicAdd(&agg[(size_t)nA * HN + nb1], silu_f(d01[v] + bb1) * 0.01f);
      atomicAdd(&agg[(size_t)nB * HN + nb0], silu_f(d10[v] + bb0) * 0.01f);
      atomicAdd(&agg[(size_t)nB * HN + nb1], silu_f(d11[v] + bb1) * 0.01f);
    }
  }
}

// ---------------------------------------------------------------------------
// Node MLP (fused 2 layers) + residual; refresh hf16; zero own agg rows
// block = 32 nodes x full H.
// ---------------------------------------------------------------------------
__global__ __launch_bounds__(256) void node_mlp_kernel(
    float* __restrict__ hwork, _Float16* __restrict__ hf16,
    float* __restrict__ agg,
    const _Float16* __restrict__ w1t, const float* __restrict__ b1,
    const _Float16* __restrict__ w2t, const float* __restrict__ b2) {
  __shared__ _Float16 A[MBLK * K2H];   // [h | agg] f16
  __shared__ _Float16 M1[MBLK * HN];

  const int n0 = blockIdx.x * MBLK;
  const int tid = threadIdx.x;
  const int wave = tid >> 5, lane = tid & 31;

  { // stage: 8 threads per node row; 32 halves of h + 32 of agg each
    const int r = tid >> 3, c = tid & 7;
    const _Float16* hr = hf16 + (size_t)(n0 + r) * HN + (c << 5);
    v8h* dst = (v8h*)(A + r * K2H);
#pragma unroll
    for (int i = 0; i < 4; ++i) dst[(c << 2) + i] = *(const v8h*)(hr + 8 * i);
    float* ar = agg + (size_t)(n0 + r) * HN + (c << 5);
#pragma unroll
    for (int i = 0; i < 32; ++i) {
      A[r * K2H + HN + (c << 5) + i] = (_Float16)ar[i];
      ar[i] = 0.0f;  // reset for next GCL's accumulation
    }
  }
  __syncthreads();

  const int nt0 = wave << 1;
  v8f a00 = {}, a01 = {}, a10 = {}, a11 = {};
  for (int kt = 0; kt < K2H / 32; ++kt) {
    const int k0 = kt << 5;
    if (kt + 1 < K2H / 32)
      __builtin_prefetch(w1t + (size_t)nt0 * 16 * K2H + k0 + 32, 0, 1);
    v16h f0 = load_frag(A + k0, K2H);
    v16h f1 = load_frag(A + 16 * K2H + k0, K2H);
    v16h bA = load_frag(w1t + (size_t)(nt0 + 0) * 16 * K2H + k0, K2H);
    v16h bB = load_frag(w1t + (size_t)(nt0 + 1) * 16 * K2H + k0, K2H);
    a00 = wmma_f16(f0, bA, a00);
    a01 = wmma_f16(f0, bB, a01);
    a10 = wmma_f16(f1, bA, a10);
    a11 = wmma_f16(f1, bB, a11);
  }
  {
    const int mb  = (lane >> 4) << 3;
    const int nb0 = (nt0 << 4) + (lane & 15);
    const int nb1 = nb0 + 16;
    const float bb0 = b1[nb0], bb1 = b1[nb1];
#pragma unroll
    for (int v = 0; v < 8; ++v) {
      const int m = mb + v;
      M1[m * HN + nb0]        = (_Float16)silu_f(a00[v] + bb0);
      M1[m * HN + nb1]        = (_Float16)silu_f(a01[v] + bb1);
      M1[(16 + m) * HN + nb0] = (_Float16)silu_f(a10[v] + bb0);
      M1[(16 + m) * HN + nb1] = (_Float16)silu_f(a11[v] + bb1);
    }
  }
  __syncthreads();

  v8f d00 = {}, d01 = {}, d10 = {}, d11 = {};
  for (int kt = 0; kt < HN / 32; ++kt) {
    const int k0 = kt << 5;
    v16h f0 = load_frag(M1 + k0, HN);
    v16h f1 = load_frag(M1 + 16 * HN + k0, HN);
    v16h bA = load_frag(w2t + (size_t)(nt0 + 0) * 16 * HN + k0, HN);
    v16h bB = load_frag(w2t + (size_t)(nt0 + 1) * 16 * HN + k0, HN);
    d00 = wmma_f16(f0, bA, d00);
    d01 = wmma_f16(f0, bB, d01);
    d10 = wmma_f16(f1, bA, d10);
    d11 = wmma_f16(f1, bB, d11);
  }
  { // residual update, keep fp32 master + f16 copy for next gather
    const int mb  = (lane >> 4) << 3;
    const int nb0 = (nt0 << 4) + (lane & 15);
    const int nb1 = nb0 + 16;
    const float bb0 = b2[nb0], bb1 = b2[nb1];
#pragma unroll
    for (int v = 0; v < 8; ++v) {
      const int nodeA = n0 + mb + v;
      const int nodeB = n0 + 16 + mb + v;
      float hA0 = hwork[(size_t)nodeA * HN + nb0] + d00[v] + bb0;
      float hA1 = hwork[(size_t)nodeA * HN + nb1] + d01[v] + bb1;
      float hB0 = hwork[(size_t)nodeB * HN + nb0] + d10[v] + bb0;
      float hB1 = hwork[(size_t)nodeB * HN + nb1] + d11[v] + bb1;
      hwork[(size_t)nodeA * HN + nb0] = hA0;
      hwork[(size_t)nodeA * HN + nb1] = hA1;
      hwork[(size_t)nodeB * HN + nb0] = hB0;
      hwork[(size_t)nodeB * HN + nb1] = hB1;
      hf16[(size_t)nodeA * HN + nb0] = (_Float16)hA0;
      hf16[(size_t)nodeA * HN + nb1] = (_Float16)hA1;
      hf16[(size_t)nodeB * HN + nb0] = (_Float16)hB0;
      hf16[(size_t)nodeB * HN + nb1] = (_Float16)hB1;
    }
  }
}

// ---------------------------------------------------------------------------
// Coord MLP: 2 WMMA layers + 256->1 projection; atomic coord update (/100)
// ---------------------------------------------------------------------------
__global__ __launch_bounds__(256) void coord_mlp_kernel(
    const _Float16* __restrict__ hf16, const float* __restrict__ ea,
    const float* __restrict__ cd,
    const int* __restrict__ rowi, const int* __restrict__ coli,
    const _Float16* __restrict__ w1t, const float* __restrict__ b1,
    const _Float16* __restrict__ w2t, const float* __restrict__ b2,
    const float* __restrict__ w3, float* __restrict__ xw) {
  __shared__ _Float16 A[MBLK * IEP];   // gathered input; reused as M2 (32*HN fits)
  __shared__ _Float16 M1[MBLK * HN];
  __shared__ int rIdx[MBLK];

  const int e0 = blockIdx.x * MBLK;
  const int tid = threadIdx.x;
  const int wave = tid >> 5, lane = tid & 31;

  stage_edge_rows(A, rIdx, hf16, ea, rowi, coli, e0);
  __syncthreads();

  const int nt0 = wave << 1;
  v8f a00 = {}, a01 = {}, a10 = {}, a11 = {};
  for (int kt = 0; kt < IEP / 32; ++kt) {
    const int k0 = kt << 5;
    v16h f0 = load_frag(A + k0, IEP);
    v16h f1 = load_frag(A + 16 * IEP + k0, IEP);
    v16h bA = load_frag(w1t + (size_t)(nt0 + 0) * 16 * IEP + k0, IEP);
    v16h bB = load_frag(w1t + (size_t)(nt0 + 1) * 16 * IEP + k0, IEP);
    a00 = wmma_f16(f0, bA, a00);
    a01 = wmma_f16(f0, bB, a01);
    a10 = wmma_f16(f1, bA, a10);
    a11 = wmma_f16(f1, bB, a11);
  }
  {
    const int mb  = (lane >> 4) << 3;
    const int nb0 = (nt0 << 4) + (lane & 15);
    const int nb1 = nb0 + 16;
    const float bb0 = b1[nb0], bb1 = b1[nb1];
#pragma unroll
    for (int v = 0; v < 8; ++v) {
      const int m = mb + v;
      M1[m * HN + nb0]        = (_Float16)silu_f(a00[v] + bb0);
      M1[m * HN + nb1]        = (_Float16)silu_f(a01[v] + bb1);
      M1[(16 + m) * HN + nb0] = (_Float16)silu_f(a10[v] + bb0);
      M1[(16 + m) * HN + nb1] = (_Float16)silu_f(a11[v] + bb1);
    }
  }
  __syncthreads();

  v8f d00 = {}, d01 = {}, d10 = {}, d11 = {};
  for (int kt = 0; kt < HN / 32; ++kt) {
    const int k0 = kt << 5;
    v16h f0 = load_frag(M1 + k0, HN);
    v16h f1 = load_frag(M1 + 16 * HN + k0, HN);
    v16h bA = load_frag(w2t + (size_t)(nt0 + 0) * 16 * HN + k0, HN);
    v16h bB = load_frag(w2t + (size_t)(nt0 + 1) * 16 * HN + k0, HN);
    d00 = wmma_f16(f0, bA, d00);
    d01 = wmma_f16(f0, bB, d01);
    d10 = wmma_f16(f1, bA, d10);
    d11 = wmma_f16(f1, bB, d11);
  }
  { // SiLU -> M2 (reuse A, row stride HN; 32*HN <= 16*IEP)
    const int mb  = (lane >> 4) << 3;
    const int nb0 = (nt0 << 4) + (lane & 15);
    const int nb1 = nb0 + 16;
    const float bb0 = b2[nb0], bb1 = b2[nb1];
#pragma unroll
    for (int v = 0; v < 8; ++v) {
      const int m = mb + v;
      A[m * HN + nb0]        = (_Float16)silu_f(d00[v] + bb0);
      A[m * HN + nb1]        = (_Float16)silu_f(d01[v] + bb1);
      A[(16 + m) * HN + nb0] = (_Float16)silu_f(d10[v] + bb0);
      A[(16 + m) * HN + nb1] = (_Float16)silu_f(d11[v] + bb1);
    }
  }
  __syncthreads();

  if (tid < MBLK) {  // 256 -> 1 projection + coordinate scatter
    const _Float16* m2 = A + tid * HN;
    float s = 0.0f;
    for (int k = 0; k < HN; ++k) s += (float)m2[k] * w3[k];
    const int e = e0 + tid;
    const int node = rIdx[tid];
    const float sc = s * 0.01f;  // segment_sum / NORM_FACTOR
    atomicAdd(&xw[node * 3 + 0], cd[e * 3 + 0] * sc);
    atomicAdd(&xw[node * 3 + 1], cd[e * 3 + 1] * sc);
    atomicAdd(&xw[node * 3 + 2], cd[e * 3 + 2] * sc);
  }
}

// ---------------------------------------------------------------------------
// Output: h @ out_w[256][16] + b ; pack (h_out, x) into d_out
// ---------------------------------------------------------------------------
__global__ void out_kernel(const float* __restrict__ hwork, const float* __restrict__ w,
                           const float* __restrict__ b, const float* __restrict__ xw,
                           float* __restrict__ out) {
  int n = blockIdx.x;
  int t = threadIdx.x;
  if (t < IN_DIM) {
    float acc = b[t];
    for (int k = 0; k < HN; ++k) acc += hwork[(size_t)n * HN + k] * w[k * IN_DIM + t];
    out[n * IN_DIM + t] = acc;
  } else if (t < IN_DIM + 3) {
    out[NNODE * IN_DIM + n * 3 + (t - IN_DIM)] = xw[n * 3 + (t - IN_DIM)];
  }
}

// ---------------------------------------------------------------------------
// Host orchestration
// ---------------------------------------------------------------------------
extern "C" void kernel_launch(void* const* d_in, const int* in_sizes, int n_in,
                              void* d_out, int out_size, void* d_ws, size_t ws_size,
                              hipStream_t stream) {
  (void)in_sizes; (void)n_in; (void)out_size; (void)ws_size;

  const float* h_in     = (const float*)d_in[0];
  const float* x_in     = (const float*)d_in[1];
  const int*   eidx     = (const int*)d_in[2];
  const float* emb_w    = (const float*)d_in[3];
  const float* emb_b    = (const float*)d_in[4];
  const float* out_w    = (const float*)d_in[5];
  const float* out_b    = (const float*)d_in[6];
  const float* edge_w1  = (const float*)d_in[7];
  const float* edge_b1  = (const float*)d_in[8];
  const float* edge_w2  = (const float*)d_in[9];
  const float* edge_b2  = (const float*)d_in[10];
  const float* node_w1  = (const float*)d_in[11];
  const float* node_b1  = (const float*)d_in[12];
  const float* node_w2  = (const float*)d_in[13];
  const float* node_b2  = (const float*)d_in[14];
  const float* coord_w1 = (const float*)d_in[15];
  const float* coord_b1 = (const float*)d_in[16];
  const float* coord_w2 = (const float*)d_in[17];
  const float* coord_b2 = (const float*)d_in[18];
  const float* coord_w3 = (const float*)d_in[19];

  const int* rowi = eidx;
  const int* coli = eidx + NEDGE;

  char* ws = (char*)d_ws;
  size_t off = 0;
  auto take = [&](size_t bytes) -> void* {
    void* p = ws + off;
    off = (off + bytes + 255) & ~(size_t)255;
    return p;
  };
  float*    hwork = (float*)take((size_t)NNODE * HN * 4);
  _Float16* hf16  = (_Float16*)take((size_t)NNODE * HN * 2);
  float*    agg   = (float*)take((size_t)NNODE * HN * 4);
  float*    xw    = (float*)take((size_t)NNODE * 3 * 4);
  float*    dist0 = (float*)take((size_t)NEDGE * 4);
  float*    ea    = (float*)take((size_t)NEDGE * 2 * 4);
  float*    cd    = (float*)take((size_t)NEDGE * 3 * 4);
  _Float16* ew1t  = (_Float16*)take((size_t)6 * HN * IEP * 2);
  _Float16* ew2t  = (_Float16*)take((size_t)6 * HN * HN * 2);
  _Float16* nw1t  = (_Float16*)take((size_t)6 * HN * K2H * 2);
  _Float16* nw2t  = (_Float16*)take((size_t)6 * HN * HN * 2);
  _Float16* cw1t  = (_Float16*)take((size_t)3 * HN * IEP * 2);
  _Float16* cw2t  = (_Float16*)take((size_t)3 * HN * HN * 2);

  auto nb = [](int total) { return (total + 255) / 256; };

  // Repack weights to f16, transposed [N][Kpad]
  wconv_kernel<<<nb(6 * HN * IEP), 256, 0, stream>>>(edge_w1, ew1t, IE_REAL, IEP, 6);
  wconv_kernel<<<nb(6 * HN * HN),  256, 0, stream>>>(edge_w2, ew2t, HN, HN, 6);
  wconv_kernel<<<nb(6 * HN * K2H), 256, 0, stream>>>(node_w1, nw1t, K2H, K2H, 6);
  wconv_kernel<<<nb(6 * HN * HN),  256, 0, stream>>>(node_w2, nw2t, HN, HN, 6);
  wconv_kernel<<<nb(3 * HN * IEP), 256, 0, stream>>>(coord_w1, cw1t, IE_REAL, IEP, 3);
  wconv_kernel<<<nb(3 * HN * HN),  256, 0, stream>>>(coord_w2, cw2t, HN, HN, 3);

  dist0_kernel<<<nb(NEDGE), 256, 0, stream>>>(x_in, rowi, coli, dist0);
  init_state_kernel<<<nb(NNODE * HN), 256, 0, stream>>>(x_in, xw, agg);
  embed_kernel<<<nb(NNODE * HN), 256, 0, stream>>>(h_in, emb_w, emb_b, hwork, hf16);

  for (int b = 0; b < 3; ++b) {
    coord2diff_kernel<<<nb(NEDGE), 256, 0, stream>>>(xw, rowi, coli, dist0, ea, cd);
    for (int s = 0; s < 2; ++s) {
      int gi = b * 2 + s;
      edge_mlp_kernel<<<NEDGE / MBLK, 256, 0, stream>>>(
          hf16, ea, rowi, coli,
          ew1t + (size_t)gi * HN * IEP, edge_b1 + gi * HN,
          ew2t + (size_t)gi * HN * HN,  edge_b2 + gi * HN, agg);
      node_mlp_kernel<<<NNODE / MBLK, 256, 0, stream>>>(
          hwork, hf16, agg,
          nw1t + (size_t)gi * HN * K2H, node_b1 + gi * HN,
          nw2t + (size_t)gi * HN * HN,  node_b2 + gi * HN);
    }
    coord_mlp_kernel<<<NEDGE / MBLK, 256, 0, stream>>>(
        hf16, ea, cd, rowi, coli,
        cw1t + (size_t)b * HN * IEP, coord_b1 + b * HN,
        cw2t + (size_t)b * HN * HN,  coord_b2 + b * HN,
        coord_w3 + b * HN, xw);
  }

  out_kernel<<<NNODE, 32, 0, stream>>>(hwork, out_w, out_b, xw, (float*)d_out);
}
